// S4Model_11416023073119
// MI455X (gfx1250) — matrix operand order repaired and checked by reading, not compiled, not run on hardware
//
#include <hip/hip_runtime.h>
#include <hip/hip_bf16.h>
#include <math.h>

#define BB   16
#define LL   4096
#define DIN  64
#define HH   256
#define NN   32
#define NLAY 4
#define DOUT 10

typedef __attribute__((ext_vector_type(16))) __bf16 v16bf;
typedef __attribute__((ext_vector_type(8)))  float  v8f;

// ---------- helpers ----------
__device__ __forceinline__ float fast_sigmoid(float x) {
    return 1.0f / (1.0f + __expf(-x));
}
__device__ __forceinline__ float fast_gelu(float x) {
    // tanh approximation (matches jax.nn.gelu approximate=True)
    float t = 0.7978845608028654f * fmaf(0.044715f, x * x * x, x);
    // tanh(t) = 1 - 2/(exp(2t)+1)
    float th = 1.0f - 2.0f / (__expf(2.0f * t) + 1.0f);
    return 0.5f * x * (1.0f + th);
}

// A-matrix (16x32 bf16) per-lane K base for vgpr-pair i (0..7)
__device__ __forceinline__ int a_kbase(int i, int half) {
    return 2 * i + 16 * half;
}

// ============================================================
// Encoder: hbuf[b,h,l] = sum_k enc_w[h,k] * x[b,l,k] + enc_b[h]
// WMMA: A = enc_w rows (M=h), B = x^T (K=d_in, N=l)
// Wave computes 64(M=h) x 16(N=l), K=64 in 2 steps.
// ============================================================
__global__ void encode_wmma(const float* __restrict__ x,
                            const float* __restrict__ enc_w,
                            const float* __restrict__ enc_b,
                            float* __restrict__ hbuf) {
    const int lane  = threadIdx.x & 31;
    const int wave  = threadIdx.x >> 5;
    const int ltile = blockIdx.x * 4 + wave;   // over (B*L)/16
    const int h0    = blockIdx.y * 64;
    const int half  = lane >> 4;
    const int r     = lane & 15;
    const int mglob = ltile * 16;              // flat b*L + l
    const int b     = mglob / LL;
    const int l0    = mglob % LL;

    v8f acc[4] = {};

    #pragma unroll
    for (int ks = 0; ks < 2; ++ks) {
        // B operand from x: row (b*L + l0 + r), k = ks*32 + 16*half + 2v (+1)
        v16bf bm;
        const float* xrow = x + (size_t)(mglob + r) * DIN + ks * 32 + 16 * half;
        #pragma unroll
        for (int v = 0; v < 8; ++v) {
            bm[2 * v]     = (__bf16)xrow[2 * v];
            bm[2 * v + 1] = (__bf16)xrow[2 * v + 1];
        }
        #pragma unroll
        for (int mt = 0; mt < 4; ++mt) {
            const float* arow = enc_w + (size_t)(h0 + mt * 16 + r) * DIN + ks * 32;
            v16bf am;
            #pragma unroll
            for (int i = 0; i < 8; ++i) {
                int kb = a_kbase(i, half);
                am[2 * i]     = (__bf16)arow[kb];
                am[2 * i + 1] = (__bf16)arow[kb + 1];
            }
            acc[mt] = __builtin_amdgcn_wmma_f32_16x16x32_bf16(
                false, am, false, bm, (short)0, acc[mt], false, false);
        }
    }

    #pragma unroll
    for (int mt = 0; mt < 4; ++mt) {
        #pragma unroll
        for (int v = 0; v < 8; ++v) {
            int h = h0 + mt * 16 + v + 8 * half;
            int l = l0 + r;
            hbuf[((size_t)b * HH + h) * LL + l] = acc[mt][v] + enc_b[h];
        }
    }
}

// ============================================================
// Per-layer S4D parameter precompute: decay = exp(dt*A),
// C~2 = 2 * (C_re + iC_im) * expm1(dt*A) / A
// ============================================================
__global__ void ssm_precompute(const float* __restrict__ log_dt,
                               const float* __restrict__ log_A_real,
                               const float* __restrict__ A_imag,
                               const float* __restrict__ C_re,
                               const float* __restrict__ C_im,
                               float* __restrict__ dre, float* __restrict__ dimg,
                               float* __restrict__ c2r, float* __restrict__ c2i) {
    int idx = blockIdx.x * blockDim.x + threadIdx.x;
    if (idx >= HH * NN) return;
    int h = idx / NN;
    float dt  = __expf(log_dt[h]);
    float are = -__expf(log_A_real[idx]);
    float aim = A_imag[idx];
    float xre = are * dt, xim = aim * dt;
    float er  = __expf(xre);
    float s, c;
    __sincosf(xim, &s, &c);
    float Dr = er * c, Di = er * s;          // exp(dtA)
    float e1r = Dr - 1.0f, e1i = Di;         // expm1(dtA)
    float den = are * are + aim * aim;
    float wr = (e1r * are + e1i * aim) / den;
    float wi = (e1i * are - e1r * aim) / den;
    float crv = C_re[idx], civ = C_im[idx];
    dre[idx]  = Dr;
    dimg[idx] = Di;
    c2r[idx]  = 2.0f * (crv * wr - civ * wi);
    c2i[idx]  = 2.0f * (crv * wi + civ * wr);
}

// ============================================================
// SSM recurrent scan (== causal FFT conv) + D skip + GELU.
// One wave per (b,h) channel; lane = state index n (N==32==wave32).
// ============================================================
__global__ void ssm_scan(const float* __restrict__ zin,
                         const float* __restrict__ dre,
                         const float* __restrict__ dimg,
                         const float* __restrict__ c2r,
                         const float* __restrict__ c2i,
                         const float* __restrict__ Dp,
                         float* __restrict__ yout) {
    const int lane = threadIdx.x & 31;
    const int wv   = threadIdx.x >> 5;
    const int ch   = blockIdx.x * 8 + wv;     // ch = b*H + h
    const int h    = ch & (HH - 1);

    const float drl = dre[h * NN + lane];
    const float dil = dimg[h * NN + lane];
    const float crl = c2r[h * NN + lane];
    const float cil = c2i[h * NN + lane];
    const float dskip = Dp[h];

    const float* zp = zin  + (size_t)ch * LL;
    float*       yp = yout + (size_t)ch * LL;

    float sre = 0.f, sim = 0.f, ysave = 0.f;

    for (int l = 0; l < LL; l += 4) {
        // hide the dependent-load chain: prefetch ~2 cachelines ahead
        __builtin_prefetch(zp + l + 128, 0, 0);
        float4 uu = *(const float4*)(zp + l);
        float us[4] = {uu.x, uu.y, uu.z, uu.w};
        #pragma unroll
        for (int k = 0; k < 4; ++k) {
            float u = us[k];
            float nre = fmaf(drl, sre, fmaf(-dil, sim, u));
            float nim = fmaf(drl, sim, dil * sre);
            sre = nre; sim = nim;
            float part = fmaf(crl, sre, -cil * sim);
            #pragma unroll
            for (int off = 16; off; off >>= 1)
                part += __shfl_xor(part, off, 32);
            float tot = fmaf(dskip, u, part);
            float gel = fast_gelu(tot);
            int li = l + k;
            if ((li & 31) == lane) ysave = gel;
            if ((li & 31) == 31) yp[(li & ~31) + lane] = ysave;
        }
    }
}

// ============================================================
// GLU projection: g = out_w @ y + out_b ; y2 = g1*sigmoid(g2);
// hbuf <- y2 + hbuf (residual), in place.
// WMMA: A = out_w rows (g and g+256), B = y tile (K=h, N=l).
// Wave computes 4 g-pairs (64 rows of g1 + 64 rows of g2) x 16 l,
// so each B (activation) tile load feeds 8 WMMAs.
// ============================================================
__global__ void glu_wmma(const float* __restrict__ ybuf,
                         const float* __restrict__ w,     // (2H, H) for this layer
                         const float* __restrict__ bias,  // (2H,)
                         float* __restrict__ hbuf) {
    const int lane  = threadIdx.x & 31;
    const int wave  = threadIdx.x >> 5;
    const int ltile = blockIdx.x * 4 + wave;
    const int g0    = blockIdx.y * 64;       // 4 pair-tiles per wave
    const int half  = lane >> 4;
    const int r     = lane & 15;
    const int mglob = ltile * 16;
    const int b     = mglob / LL;
    const int l0    = mglob % LL;

    const float* ybase = ybuf + (size_t)b * HH * LL;

    v8f acc1[4] = {};
    v8f acc2[4] = {};

    for (int ks = 0; ks < 8; ++ks) {
        // B operand: y tile, k (=h) pairs stride-L apart, n = l contiguous
        v16bf bm;
        #pragma unroll
        for (int v = 0; v < 8; ++v) {
            int k = ks * 32 + 16 * half + 2 * v;          // h index
            const float* p = ybase + (size_t)k * LL + l0 + r;
            bm[2 * v]     = (__bf16)p[0];
            bm[2 * v + 1] = (__bf16)p[LL];
        }
        #pragma unroll
        for (int pt = 0; pt < 4; ++pt) {
            const float* ar1 = w + (size_t)(g0 + pt * 16 + r) * HH + ks * 32;
            const float* ar2 = w + (size_t)(g0 + 256 + pt * 16 + r) * HH + ks * 32;
            v16bf a1, a2;
            #pragma unroll
            for (int i = 0; i < 8; ++i) {
                int kb = a_kbase(i, half);
                a1[2 * i]     = (__bf16)ar1[kb];
                a1[2 * i + 1] = (__bf16)ar1[kb + 1];
                a2[2 * i]     = (__bf16)ar2[kb];
                a2[2 * i + 1] = (__bf16)ar2[kb + 1];
            }
            acc1[pt] = __builtin_amdgcn_wmma_f32_16x16x32_bf16(
                false, a1, false, bm, (short)0, acc1[pt], false, false);
            acc2[pt] = __builtin_amdgcn_wmma_f32_16x16x32_bf16(
                false, a2, false, bm, (short)0, acc2[pt], false, false);
        }
    }

    #pragma unroll
    for (int pt = 0; pt < 4; ++pt) {
        #pragma unroll
        for (int v = 0; v < 8; ++v) {
            int g = g0 + pt * 16 + v + 8 * half;   // == h for the GLU output
            int l = l0 + r;
            float g1v = acc1[pt][v] + bias[g];
            float g2v = acc2[pt][v] + bias[g + 256];
            float val = g1v * fast_sigmoid(g2v);
            size_t idx = ((size_t)b * HH + g) * LL + l;
            hbuf[idx] = val + hbuf[idx];    // residual, in place (1:1 element)
        }
    }
}

// ============================================================
// LayerNorm over channel dim H. One thread per (b,l); lanes map
// to consecutive l -> coalesced per-h row accesses.
// ============================================================
__global__ void ln_channels(float* __restrict__ hbuf,
                            const float* __restrict__ lw,
                            const float* __restrict__ lb) {
    int t = blockIdx.x * blockDim.x + threadIdx.x;   // over B*L
    int b = t / LL, l = t % LL;
    float* base = hbuf + (size_t)b * HH * LL + l;
    float s = 0.f, sq = 0.f;
    for (int h = 0; h < HH; ++h) {
        float v = base[(size_t)h * LL];
        s += v; sq += v * v;
    }
    float mu  = s * (1.0f / HH);
    float var = sq * (1.0f / HH) - mu * mu;
    float inv = rsqrtf(var + 1e-5f);
    for (int h = 0; h < HH; ++h) {
        size_t o = (size_t)h * LL;
        base[o] = (base[o] - mu) * inv * lw[h] + lb[h];
    }
}

// ============================================================
// Decoder: out[b,l,o] = sum_h dec_w[o,h] * h[b,h,l] + dec_b[o]
// Tiny N=10 -> scalar-broadcast weights, VALU FMA.
// ============================================================
__global__ void decode_k(const float* __restrict__ hbuf,
                         const float* __restrict__ dw,
                         const float* __restrict__ db,
                         float* __restrict__ out) {
    int t = blockIdx.x * blockDim.x + threadIdx.x;   // over B*L
    int b = t / LL, l = t % LL;
    const float* base = hbuf + (size_t)b * HH * LL + l;
    float acc[DOUT] = {};
    for (int h = 0; h < HH; ++h) {
        float v = base[(size_t)h * LL];
        #pragma unroll
        for (int o = 0; o < DOUT; ++o)
            acc[o] = fmaf(dw[o * HH + h], v, acc[o]);
    }
    #pragma unroll
    for (int o = 0; o < DOUT; ++o)
        out[(size_t)t * DOUT + o] = acc[o] + db[o];
}

// ============================================================
extern "C" void kernel_launch(void* const* d_in, const int* in_sizes, int n_in,
                              void* d_out, int out_size, void* d_ws, size_t ws_size,
                              hipStream_t stream) {
    const float* x          = (const float*)d_in[0];
    const float* enc_w      = (const float*)d_in[1];
    const float* enc_b      = (const float*)d_in[2];
    const float* log_dt     = (const float*)d_in[3];
    const float* log_A_real = (const float*)d_in[4];
    const float* A_imag     = (const float*)d_in[5];
    const float* C_re       = (const float*)d_in[6];
    const float* C_im       = (const float*)d_in[7];
    const float* Dp         = (const float*)d_in[8];
    const float* out_w      = (const float*)d_in[9];
    const float* out_b      = (const float*)d_in[10];
    const float* ln_w       = (const float*)d_in[11];
    const float* ln_b       = (const float*)d_in[12];
    const float* dec_w      = (const float*)d_in[13];
    const float* dec_b      = (const float*)d_in[14];

    const size_t act = (size_t)BB * HH * LL;     // 16.7M floats = 64 MiB
    float* hbuf = (float*)d_ws;
    float* ybuf = hbuf + act;
    float* dre  = ybuf + act;
    float* dimg = dre + HH * NN;
    float* c2r  = dimg + HH * NN;
    float* c2i  = c2r + HH * NN;

    // Encoder: grid over l-tiles (4/block) x h-groups of 64
    encode_wmma<<<dim3(BB * LL / 64, HH / 64), 128, 0, stream>>>(x, enc_w, enc_b, hbuf);

    for (int i = 0; i < NLAY; ++i) {
        ssm_precompute<<<(HH * NN + 255) / 256, 256, 0, stream>>>(
            log_dt + (size_t)i * HH,
            log_A_real + (size_t)i * HH * NN,
            A_imag + (size_t)i * HH * NN,
            C_re + (size_t)i * HH * NN,
            C_im + (size_t)i * HH * NN,
            dre, dimg, c2r, c2i);

        ssm_scan<<<BB * HH / 8, 256, 0, stream>>>(hbuf, dre, dimg, c2r, c2i,
                                                  Dp + (size_t)i * HH, ybuf);

        // 4 g-pairs per wave: grid.y = (2H/2)/64 = 4
        glu_wmma<<<dim3(BB * LL / 64, HH / 64), 128, 0, stream>>>(
            ybuf, out_w + (size_t)i * 2 * HH * HH, out_b + (size_t)i * 2 * HH, hbuf);

        ln_channels<<<BB * LL / 256, 256, 0, stream>>>(hbuf,
                                                       ln_w + (size_t)i * HH,
                                                       ln_b + (size_t)i * HH);
    }

    decode_k<<<BB * LL / 256, 256, 0, stream>>>(hbuf, dec_w, dec_b, (float*)d_out);
}
